// Transformer_19430432047158
// MI455X (gfx1250) — matrix operand order repaired and checked by reading, not compiled, not run on hardware
//
#include <hip/hip_runtime.h>
#include <hip/hip_bf16.h>

typedef __attribute__((ext_vector_type(16))) __bf16 v16bf;
typedef __attribute__((ext_vector_type(8)))  __bf16 v8bf;
typedef __attribute__((ext_vector_type(4)))  __bf16 v4bf;
typedef __attribute__((ext_vector_type(8)))  float  v8f;

#define DEV static __device__ __forceinline__

constexpr int kSeq = 8192;
constexpr int kD   = 512;
constexpr int kFF  = 2048;

union Frag16 { v16bf v; v8bf h[2]; };

DEV v8f wmma_bf16(v16bf a, v16bf b, v8f c) {
  // D = A(16x32 bf16) * B(32x16 bf16) + C(16x16 f32)
  return __builtin_amdgcn_wmma_f32_16x16x32_bf16(false, a, false, b, (short)0, c,
                                                 false, false);
}

// CDNA5 async copy: 16B global -> LDS, no VGPR transit, tracked by ASYNCcnt.
// LDS operand = low 32 bits of the generic pointer (ISA 10.2: LDS aperture
// truncates flat address to addr[31:0]).
DEV void async_ld16(__bf16* lds, const __bf16* g) {
  asm volatile("global_load_async_to_lds_b128 %0, %1, off"
               :
               : "v"((unsigned)(uintptr_t)lds), "v"(g)
               : "memory");
}
DEV void wait_async0() { asm volatile("s_wait_asynccnt 0" ::: "memory"); }

// A fragment (16x32, row-major tile in LDS). ISA layout:
// lane<16 : row=lane, elems 0-7 = K k0..k0+7, elems 8-15 = K k0+16..k0+23
// lane>=16: row=lane-16, K halves shifted by +8.
DEV v16bf ld_a_frag(const __bf16* tile, int ld, int k0) {
  const int lane = threadIdx.x & 31;
  const int row  = lane & 15;
  const int sel  = (lane >> 4) << 3;
  const __bf16* p = tile + row * ld + k0 + sel;
  Frag16 f;
  f.h[0] = *(const v8bf*)(p);
  f.h[1] = *(const v8bf*)(p + 16);
  return f.v;
}

// B fragment (32x16) from N-major (transposed) LDS tile:
// lane<16 : col=lane, elems 0-7 = K k0..k0+7, elems 8-15 = K k0+16..k0+23; lane>=16: col=lane-16, K halves shifted by +8.
DEV v16bf ld_b_frag(const __bf16* tileT, int ld, int n0, int k0) {
  const int lane = threadIdx.x & 31;
  const int col  = lane & 15;
  const int kg   = (lane >> 4) << 3;
  const __bf16* p = tileT + (n0 + col) * ld + k0 + kg;
  Frag16 f;
  f.h[0] = *(const v8bf*)(p);
  f.h[1] = *(const v8bf*)(p + 16);
  return f.v;
}

// ---------------------------------------------------------------------------
// Embedding + positional encoding: x1 = emb[x] + PE  (f32 + bf16 copies)
// ---------------------------------------------------------------------------
__global__ __launch_bounds__(256)
void embed_pe_kernel(const int* __restrict__ x, const float* __restrict__ emb,
                     float* __restrict__ x1f, __bf16* __restrict__ x1b)
{
  const int i = blockIdx.x * 256 + threadIdx.x;
  const int s = i >> 9;
  const int d = i & (kD - 1);
  float v = emb[(size_t)x[s] * kD + d];
  const float ang =
      (float)s * __expf(-(2.0f * (float)d / (float)kD) * 9.21034037197618f);
  v += (d & 1) ? cosf(ang) : sinf(ang);
  x1f[i] = v;
  x1b[i] = (__bf16)v;
}

// ---------------------------------------------------------------------------
// Generic WMMA GEMM: C[M,N] = act(A_bf16[M,K] @ B_f32[K,N] + bias (+resid))
// 256 threads, 64x128 block tile, BK=64, wave -> 32x32 accumulator.
// A tile staged with async global->LDS; B tile converted f32->bf16 + transposed.
// ---------------------------------------------------------------------------
__global__ __launch_bounds__(256)
void gemm_bf16_wmma(const __bf16* __restrict__ A, const float* __restrict__ B,
                    const float* __restrict__ bias, const float* __restrict__ resid,
                    float* __restrict__ outF, __bf16* __restrict__ outB,
                    int M, int N, int K, int relu)
{
  constexpr int LDA = 72;                 // 64 + 8 pad (16B aligned rows)
  __shared__ __bf16 At[64 * LDA];         // A tile, row-major
  __shared__ __bf16 Bt[128 * LDA];        // B tile, N-major (transposed)
  const int tid  = threadIdx.x;
  const int lane = tid & 31;
  const int wave = tid >> 5;
  const int mw = (wave & 1) << 5;
  const int nw = (wave >> 1) << 5;
  const int m0 = blockIdx.x << 6;
  const int n0 = blockIdx.y << 7;

  const v8f vz = {0.f,0.f,0.f,0.f,0.f,0.f,0.f,0.f};
  v8f acc[4];
  acc[0] = vz; acc[1] = vz; acc[2] = vz; acc[3] = vz;

  for (int k0 = 0; k0 < K; k0 += 64) {
    // stage A (64x64 bf16): async 16B copies straight into LDS
#pragma unroll
    for (int i = 0; i < 2; ++i) {
      const int c   = tid + (i << 8);
      const int row = c >> 3;
      const int kc  = (c & 7) << 3;
      async_ld16(At + row * LDA + kc, A + (size_t)(m0 + row) * K + k0 + kc);
    }
    // stage B (64x128 f32) transposed -> bf16 N-major; prefetch next tile
#pragma unroll
    for (int i = 0; i < 8; ++i) {
      const int idx = tid + (i << 8);
      const int kr  = idx >> 5;
      const int nc  = (idx & 31) << 2;
      const float* bp = B + (size_t)(k0 + kr) * N + n0 + nc;
      const float4 bv = *(const float4*)bp;
      if (k0 + 64 < K) __builtin_prefetch(bp + (size_t)64 * N, 0, 1);
      Bt[(nc + 0) * LDA + kr] = (__bf16)bv.x;
      Bt[(nc + 1) * LDA + kr] = (__bf16)bv.y;
      Bt[(nc + 2) * LDA + kr] = (__bf16)bv.z;
      Bt[(nc + 3) * LDA + kr] = (__bf16)bv.w;
    }
    wait_async0();
    __syncthreads();
#pragma unroll
    for (int kk = 0; kk < 64; kk += 32) {
      v16bf a0 = ld_a_frag(At + (mw +  0) * LDA, LDA, kk);
      v16bf a1 = ld_a_frag(At + (mw + 16) * LDA, LDA, kk);
      v16bf b0 = ld_b_frag(Bt, LDA, nw +  0, kk);
      v16bf b1 = ld_b_frag(Bt, LDA, nw + 16, kk);
      acc[0] = wmma_bf16(a0, b0, acc[0]);
      acc[1] = wmma_bf16(a0, b1, acc[1]);
      acc[2] = wmma_bf16(a1, b0, acc[2]);
      acc[3] = wmma_bf16(a1, b1, acc[3]);
    }
    __syncthreads();
  }

  const int colL = lane & 15;
  const int rowH = (lane >> 4) << 3;   // C layout: row = rowH + r, col = colL
#pragma unroll
  for (int t = 0; t < 4; ++t) {
    const int ti = t >> 1, tj = t & 1;
    const int col = n0 + nw + (tj << 4) + colL;
    const float bi = bias[col];
#pragma unroll
    for (int r = 0; r < 8; ++r) {
      const int row = m0 + mw + (ti << 4) + rowH + r;
      float v = acc[t][r] + bi;
      if (resid) v += resid[(size_t)row * N + col];
      if (relu)  v = fmaxf(v, 0.f);
      const size_t o = (size_t)row * N + col;
      if (outF) outF[o] = v;
      if (outB) outB[o] = (__bf16)v;
    }
  }
}

// ---------------------------------------------------------------------------
// Attention pass 1: per-row softmax stats (m_i, l_i) via streaming WMMA QK^T.
// 32 query rows / block, 64-key tiles, 8 waves = 2(M) x 4(key-sub).
// ---------------------------------------------------------------------------
__global__ __launch_bounds__(256)
void attn_stats_kernel(const __bf16* __restrict__ Q, const __bf16* __restrict__ Kmat,
                       float* __restrict__ stats)
{
  constexpr int LQ = 520;               // 512 + 8 pad
  extern __shared__ __bf16 smem[];
  __bf16* Qt = smem;                    // 32 x LQ
  __bf16* Kt = smem + 32 * LQ;          // 64 x LQ (row-major K == N-major K^T)
  float* pm = (float*)(smem + (32 + 64) * LQ);   // [8][16]
  float* pl = pm + 8 * 16;

  const int tid  = threadIdx.x;
  const int lane = tid & 31;
  const int wave = tid >> 5;
  const int ms = (wave & 1) << 4;
  const int ns = (wave >> 1) << 4;
  const int m0 = blockIdx.x << 5;
  const float scale = 0.0441941738241592f;   // 1/sqrt(512)

  for (int i = tid; i < 32 * 64; i += 256) {
    const int row = i >> 6, kc = (i & 63) << 3;
    async_ld16(Qt + row * LQ + kc, Q + (size_t)(m0 + row) * kD + kc);
  }

  float rm[8], rl[8];
#pragma unroll
  for (int r = 0; r < 8; ++r) { rm[r] = -3.0e38f; rl[r] = 0.f; }

  for (int j0 = 0; j0 < kSeq; j0 += 64) {
    __syncthreads();
    for (int i = tid; i < 64 * 64; i += 256) {
      const int row = i >> 6, kc = (i & 63) << 3;
      async_ld16(Kt + row * LQ + kc, Kmat + (size_t)(j0 + row) * kD + kc);
    }
    wait_async0();
    __syncthreads();
    v8f s = {0.f,0.f,0.f,0.f,0.f,0.f,0.f,0.f};
    for (int kk = 0; kk < kD; kk += 32) {
      v16bf a = ld_a_frag(Qt + ms * LQ, LQ, kk);
      v16bf b = ld_b_frag(Kt, LQ, ns, kk);
      s = wmma_bf16(a, b, s);
    }
    // online (max, sum-exp) per row; cross-lane reduce over the 16 cols
#pragma unroll
    for (int r = 0; r < 8; ++r) {
      const float v = s[r] * scale;
      float mx = v;
#pragma unroll
      for (int o = 1; o < 16; o <<= 1) mx = fmaxf(mx, __shfl_xor(mx, o, 16));
      const float mnew = fmaxf(rm[r], mx);
      float e = __expf(v - mnew);
#pragma unroll
      for (int o = 1; o < 16; o <<= 1) e += __shfl_xor(e, o, 16);
      rl[r] = rl[r] * __expf(rm[r] - mnew) + e;
      rm[r] = mnew;
    }
  }
  if ((lane & 15) == 0) {
    const int half = lane >> 4;
#pragma unroll
    for (int r = 0; r < 8; ++r) {
      pm[wave * 16 + (half << 3) + r] = rm[r];
      pl[wave * 16 + (half << 3) + r] = rl[r];
    }
  }
  __syncthreads();
  if (tid < 32) {
    const int msub = tid >> 4;
    const int rr   = tid & 15;
    float m = -3.0e38f;
#pragma unroll
    for (int w = 0; w < 4; ++w) m = fmaxf(m, pm[(msub + 2 * w) * 16 + rr]);
    float l = 0.f;
#pragma unroll
    for (int w = 0; w < 4; ++w)
      l += pl[(msub + 2 * w) * 16 + rr] * __expf(pm[(msub + 2 * w) * 16 + rr] - m);
    stats[(size_t)(m0 + tid) * 2 + 0] = m;
    stats[(size_t)(m0 + tid) * 2 + 1] = l;
  }
}

// ---------------------------------------------------------------------------
// Attention pass 2: O = softmax(QK^T/sqrt(D)) @ V using stats.
// 32 query rows / block; key tiles of 32; waves 0-3 recompute S -> P (LDS),
// all 8 waves (2M x 4 of 128 cols) accumulate P@V with WMMA.
// ---------------------------------------------------------------------------
__global__ __launch_bounds__(256)
void attn_out_kernel(const __bf16* __restrict__ Q, const __bf16* __restrict__ Kmat,
                     const __bf16* __restrict__ V, const float* __restrict__ stats,
                     __bf16* __restrict__ O)
{
  constexpr int LQ = 520;
  constexpr int LV = 40;                 // 32 + 8 pad
  extern __shared__ __bf16 smem[];
  __bf16* Qt = smem;                     // 32 x LQ
  __bf16* Kt = Qt + 32 * LQ;             // 32 x LQ
  __bf16* Vt = Kt + 32 * LQ;             // 512 x LV (V transposed: [d][k])
  __bf16* Pt = Vt + 512 * LV;            // 32 x LV

  const int tid  = threadIdx.x;
  const int lane = tid & 31;
  const int wave = tid >> 5;
  const int ms   = (wave & 1) << 4;
  const int dsub = (wave >> 1) << 7;     // 128-col slab
  const int m0   = blockIdx.x << 5;
  const float scale = 0.0441941738241592f;

  float sm[8], sli[8];
#pragma unroll
  for (int r = 0; r < 8; ++r) { sm[r] = 0.f; sli[r] = 1.f; }
  if (wave < 4) {
    const int base = m0 + ((wave & 1) << 4) + ((lane >> 4) << 3);
#pragma unroll
    for (int r = 0; r < 8; ++r) {
      sm[r]  = stats[(size_t)(base + r) * 2 + 0];
      sli[r] = 1.f / stats[(size_t)(base + r) * 2 + 1];
    }
  }

  for (int i = tid; i < 32 * 64; i += 256) {
    const int row = i >> 6, kc = (i & 63) << 3;
    async_ld16(Qt + row * LQ + kc, Q + (size_t)(m0 + row) * kD + kc);
  }

  const v8f vz = {0.f,0.f,0.f,0.f,0.f,0.f,0.f,0.f};
  v8f oacc[8];
#pragma unroll
  for (int j = 0; j < 8; ++j) oacc[j] = vz;

  for (int j0 = 0; j0 < kSeq; j0 += 32) {
    __syncthreads();
    for (int i = tid; i < 32 * 64; i += 256) {
      const int row = i >> 6, kc = (i & 63) << 3;
      async_ld16(Kt + row * LQ + kc, Kmat + (size_t)(j0 + row) * kD + kc);
    }
    for (int i = tid; i < 32 * 128; i += 256) {
      const int k  = i >> 7;
      const int d4 = (i & 127) << 2;
      const v4bf v = *(const v4bf*)(V + (size_t)(j0 + k) * kD + d4);
      Vt[(d4 + 0) * LV + k] = v[0];
      Vt[(d4 + 1) * LV + k] = v[1];
      Vt[(d4 + 2) * LV + k] = v[2];
      Vt[(d4 + 3) * LV + k] = v[3];
    }
    wait_async0();
    __syncthreads();
    if (wave < 4) {
      const int mss = (wave & 1) << 4;
      const int nss = ((wave >> 1) & 1) << 4;
      v8f s = vz;
      for (int kk = 0; kk < kD; kk += 32) {
        v16bf a = ld_a_frag(Qt + mss * LQ, LQ, kk);
        v16bf b = ld_b_frag(Kt, LQ, nss, kk);
        s = wmma_bf16(a, b, s);
      }
      const int prow = mss + ((lane >> 4) << 3);
      const int pcol = nss + (lane & 15);
#pragma unroll
      for (int r = 0; r < 8; ++r) {
        const float p = __expf(s[r] * scale - sm[r]) * sli[r];
        Pt[(prow + r) * LV + pcol] = (__bf16)p;
      }
    }
    __syncthreads();
    v16bf a = ld_a_frag(Pt + ms * LV, LV, 0);
#pragma unroll
    for (int j = 0; j < 8; ++j) {
      v16bf b = ld_b_frag(Vt, LV, dsub + (j << 4), 0);
      oacc[j] = wmma_bf16(a, b, oacc[j]);
    }
  }

  const int colL = lane & 15;
  const int rowH = (lane >> 4) << 3;
#pragma unroll
  for (int j = 0; j < 8; ++j) {
    const int col = dsub + (j << 4) + colL;
#pragma unroll
    for (int r = 0; r < 8; ++r) {
      const int row = m0 + ms + rowH + r;
      O[(size_t)row * kD + col] = (__bf16)oacc[j][r];
    }
  }
}

// ---------------------------------------------------------------------------
// Global LayerNorm: tensor-wide mean + unbiased var; eps added after sqrt.
// ---------------------------------------------------------------------------
__global__ __launch_bounds__(256)
void ln_partial_kernel(const float* __restrict__ a, const float* __restrict__ b,
                       int n, float* __restrict__ part)
{
  __shared__ float ss[256], sq[256];
  const int tid = threadIdx.x;
  float s = 0.f, q = 0.f;
  for (int i = blockIdx.x * 256 + tid; i < n; i += gridDim.x * 256) {
    const float v = a[i] + (b ? b[i] : 0.f);
    s += v; q += v * v;
  }
  ss[tid] = s; sq[tid] = q;
  __syncthreads();
  for (int o = 128; o > 0; o >>= 1) {
    if (tid < o) { ss[tid] += ss[tid + o]; sq[tid] += sq[tid + o]; }
    __syncthreads();
  }
  if (tid == 0) { part[blockIdx.x * 2] = ss[0]; part[blockIdx.x * 2 + 1] = sq[0]; }
}

__global__ __launch_bounds__(256)
void ln_final_kernel(const float* __restrict__ part, int nb, int n,
                     float* __restrict__ res)
{
  __shared__ float ss[256], sq[256];
  const int tid = threadIdx.x;
  float s = 0.f, q = 0.f;
  if (tid < nb) { s = part[tid * 2]; q = part[tid * 2 + 1]; }
  ss[tid] = s; sq[tid] = q;
  __syncthreads();
  for (int o = 128; o > 0; o >>= 1) {
    if (tid < o) { ss[tid] += ss[tid + o]; sq[tid] += sq[tid + o]; }
    __syncthreads();
  }
  if (tid == 0) {
    const float fn = (float)n;
    const float mean = ss[0] / fn;
    const float var  = (sq[0] - fn * mean * mean) / (fn - 1.f);
    res[0] = mean;
    res[1] = 1.f / (sqrtf(var) + 1e-9f);
  }
}

__global__ __launch_bounds__(256)
void ln_apply_kernel(const float* __restrict__ a, const float* __restrict__ b,
                     const float* __restrict__ gamma, const float* __restrict__ beta,
                     const float* __restrict__ res, float* __restrict__ outF,
                     __bf16* __restrict__ outB, int n)
{
  const int i = blockIdx.x * 256 + threadIdx.x;
  if (i >= n) return;
  const float mean = res[0];
  const float fac  = res[1];
  const int d = i & (kD - 1);
  const float v = a[i] + (b ? b[i] : 0.f);
  const float y = gamma[d] * ((v - mean) * fac) + beta[d];
  if (outF) outF[i] = y;
  if (outB) outB[i] = (__bf16)y;
}

// ---------------------------------------------------------------------------
extern "C" void kernel_launch(void* const* d_in, const int* in_sizes, int n_in,
                              void* d_out, int out_size, void* d_ws, size_t ws_size,
                              hipStream_t stream)
{
  (void)in_sizes; (void)n_in; (void)out_size; (void)ws_size;
  const int*   x     = (const int*)  d_in[0];
  const float* emb   = (const float*)d_in[1];
  const float* Wq    = (const float*)d_in[2];
  const float* bq    = (const float*)d_in[3];
  const float* Wk    = (const float*)d_in[4];
  const float* bk    = (const float*)d_in[5];
  const float* Wv    = (const float*)d_in[6];
  const float* bv    = (const float*)d_in[7];
  const float* Wo    = (const float*)d_in[8];
  const float* bo    = (const float*)d_in[9];
  const float* W1    = (const float*)d_in[10];
  const float* b1    = (const float*)d_in[11];
  const float* W2    = (const float*)d_in[12];
  const float* b2    = (const float*)d_in[13];
  const float* gamma = (const float*)d_in[14];
  const float* beta  = (const float*)d_in[15];
  float* out = (float*)d_out;

  char* base = (char*)d_ws;
  size_t off = 0;
  auto alloc = [&](size_t bytes) -> void* {
    void* p = base + off;
    off += (bytes + 255) & ~(size_t)255;
    return p;
  };
  const size_t SD = (size_t)kSeq * kD;
  float*  x1f   = (float*) alloc(SD * 4);          // [0, 16M)
  __bf16* x1b   = (__bf16*)alloc(SD * 2);          // [16M, 24M)
  __bf16* qb    = (__bf16*)alloc(SD * 2);          // [24M, 32M)
  __bf16* kb    = (__bf16*)alloc(SD * 2);          // [32M, 40M)
  __bf16* vb    = (__bf16*)alloc(SD * 2);          // [40M, 48M)
  __bf16* ob    = (__bf16*)alloc(SD * 2);          // [48M, 56M)
  float*  yf    = (float*) alloc(SD * 4);          // [56M, 72M)  x1+x2
  float*  x3f   = (float*) alloc(SD * 4);          // [72M, 88M)
  __bf16* x3b   = (__bf16*)alloc(SD * 2);          // [88M, 96M)
  float*  stats = (float*) alloc((size_t)kSeq * 2 * 4);
  float*  red   = (float*) alloc((256 * 2 + 2) * 4);
  float*  res   = red + 512;
  // dead-buffer aliases (lifetimes disjoint):
  __bf16* ffb = (__bf16*)x1f;   // 32MB over x1f/x1b/qb (all dead before FFN1)
  float*  zf  = (float*)kb;     // 16MB over kb/vb (dead before FFN2)

  const unsigned STATS_SMEM = (32 + 64) * 520 * 2 + 2 * 8 * 16 * 4; // ~100KB
  const unsigned OUT_SMEM   = (32 + 32) * 520 * 2 + 512 * 40 * 2 + 32 * 40 * 2;

  // 1) embedding + positional encoding
  embed_pe_kernel<<<SD / 256, 256, 0, stream>>>(x, emb, x1f, x1b);

  // 2) Q,K,V projections (bf16 outputs)
  dim3 gD(kSeq / 64, kD / 128);
  gemm_bf16_wmma<<<gD, 256, 0, stream>>>(x1b, Wq, bq, nullptr, nullptr, qb,
                                         kSeq, kD, kD, 0);
  gemm_bf16_wmma<<<gD, 256, 0, stream>>>(x1b, Wk, bk, nullptr, nullptr, kb,
                                         kSeq, kD, kD, 0);
  gemm_bf16_wmma<<<gD, 256, 0, stream>>>(x1b, Wv, bv, nullptr, nullptr, vb,
                                         kSeq, kD, kD, 0);

  // 3) flash-style attention (stats pass + output pass)
  attn_stats_kernel<<<kSeq / 32, 256, STATS_SMEM, stream>>>(qb, kb, stats);
  attn_out_kernel<<<kSeq / 32, 256, OUT_SMEM, stream>>>(qb, kb, vb, stats, ob);

  // 4) output projection + residual: yf = attn @ Wo + bo + x1
  gemm_bf16_wmma<<<gD, 256, 0, stream>>>(ob, Wo, bo, x1f, yf, nullptr,
                                         kSeq, kD, kD, 0);

  // 5) global layernorm 1 -> x3 (f32 + bf16)
  ln_partial_kernel<<<256, 256, 0, stream>>>(yf, nullptr, (int)SD, red);
  ln_final_kernel<<<1, 256, 0, stream>>>(red, 256, (int)SD, res);
  ln_apply_kernel<<<SD / 256, 256, 0, stream>>>(yf, nullptr, gamma, beta, res,
                                                x3f, x3b, (int)SD);

  // 6) FFN: relu(x3 @ W1 + b1) -> ffb (bf16), then ffb @ W2 + b2 -> zf
  dim3 gF(kSeq / 64, kFF / 128);
  gemm_bf16_wmma<<<gF, 256, 0, stream>>>(x3b, W1, b1, nullptr, nullptr, ffb,
                                         kSeq, kFF, kD, 1);
  gemm_bf16_wmma<<<gD, 256, 0, stream>>>(ffb, W2, b2, nullptr, zf, nullptr,
                                         kSeq, kD, kFF, 0);

  // 7) global layernorm 2 over (x3 + x4) -> output
  ln_partial_kernel<<<256, 256, 0, stream>>>(x3f, zf, (int)SD, red);
  ln_final_kernel<<<1, 256, 0, stream>>>(red, 256, (int)SD, res);
  ln_apply_kernel<<<SD / 256, 256, 0, stream>>>(x3f, zf, gamma, beta, res,
                                                out, nullptr, (int)SD);
}
